// Attention_5609227289010
// MI455X (gfx1250) — matrix operand order repaired and checked by reading, not compiled, not run on hardware
//
#include <hip/hip_runtime.h>

// ---------------------------------------------------------------------------
// Problem constants (from reference)
// ---------------------------------------------------------------------------
#define B_   2
#define S_   2048
#define H_   4096
#define NH_  32
#define NKV_ 8
#define HD_  128
#define G_   (NH_ / NKV_)          // 4
#define NQKV (H_ + 2 * NKV_ * HD_) // 6144
#define NEGV (-1.0e9f)
#define SCALE_ (0.08838834764831843f) // 1/sqrt(128)

typedef __attribute__((ext_vector_type(16))) __bf16 v16bf;
typedef __attribute__((ext_vector_type(8)))  float  v8f;
typedef __attribute__((ext_vector_type(4)))  unsigned int u32x4;
typedef __attribute__((ext_vector_type(8)))  int  i32x8;
typedef __attribute__((ext_vector_type(4)))  int  i32x4;

union FragAB {
    v16bf v;
    unsigned int   u[8];
    unsigned short s[16];
};

__device__ __forceinline__ unsigned short f32_to_bf16(float f) {
    union { float f; unsigned int u; } x; x.f = f;
    unsigned int r = x.u + 0x7FFFu + ((x.u >> 16) & 1u); // round-nearest-even
    return (unsigned short)(r >> 16);
}
__device__ __forceinline__ float bf16_to_f32(unsigned short u) {
    union { unsigned int i; float f; } x; x.i = ((unsigned int)u) << 16;
    return x.f;
}

// ---------------------------------------------------------------------------
// Kernel 1/4: bf16 WMMA GEMM  C[M,N] = A[M,K] * W[K,N] + bias
// Block: 256 threads (8 waves). Tile: 128x128, K-step 32.
// ---------------------------------------------------------------------------
template<bool A_IS_F32, bool OUT_F32>
__global__ __launch_bounds__(256)
void gemm_wmma_bf16(const void* __restrict__ Aptr,
                    const float* __restrict__ W,
                    const float* __restrict__ bias,
                    void* __restrict__ Cptr,
                    int M, int N, int K)
{
    __shared__ unsigned short lds_a [128][34]; // row-major [m][k], padded
    __shared__ unsigned short lds_bt[128][34]; // transposed [n][k], padded

    const int tid  = threadIdx.x;
    const int wave = tid >> 5;
    const int lane = tid & 31;
    const int lrow = lane & 15;
    const int hi   = lane >> 4;

    const int m0 = blockIdx.y * 128;
    const int n0 = blockIdx.x * 128;

    v8f acc[8];
    #pragma unroll
    for (int i = 0; i < 8; ++i) acc[i] = {};

    const float*          Af = (const float*)Aptr;
    const unsigned short* Ab = (const unsigned short*)Aptr;

    const int kTiles = K >> 5;
    for (int kt = 0; kt < kTiles; ++kt) {
        const int k0 = kt << 5;
        // ---- fill A tile (128x32) ----
        #pragma unroll
        for (int i = 0; i < 16; ++i) {
            int idx = i * 256 + tid;
            int r = idx >> 5, c = idx & 31;
            size_t g = (size_t)(m0 + r) * K + (k0 + c);
            lds_a[r][c] = A_IS_F32 ? f32_to_bf16(Af[g]) : Ab[g];
        }
        // ---- fill B tile (32x128) transposed into [n][k] ----
        #pragma unroll
        for (int i = 0; i < 16; ++i) {
            int idx = i * 256 + tid;
            int r = idx >> 7, c = idx & 127; // r = k, c = n
            lds_bt[c][r] = f32_to_bf16(W[(size_t)(k0 + r) * N + (n0 + c)]);
        }
        // ---- prefetch next K-tile (global_prefetch_b8) ----
        if (kt + 1 < kTiles) {
            int rA = tid >> 5, cA = tid & 31;
            size_t gA = (size_t)(m0 + rA) * K + (k0 + 32 + cA);
            __builtin_prefetch(A_IS_F32 ? (const void*)&Af[gA]
                                        : (const void*)&Ab[gA], 0, 1);
            int rB = tid >> 7, cB = tid & 127;
            __builtin_prefetch(&W[(size_t)(k0 + 32 + rB) * N + (n0 + cB)], 0, 1);
        }
        __syncthreads();

        // ---- A fragment: 16x32, rows m = wave*16 + lrow ----
        FragAB af;
        const int am = wave * 16 + lrow;
        #pragma unroll
        for (int j = 0; j < 8; ++j) {
            int kb = ((j & 4) << 2) + hi * 8 + (j & 3) * 2;
            af.u[j] = *(const unsigned int*)&lds_a[am][kb];
        }
        // ---- 8 WMMAs over the 128-wide N tile ----
        #pragma unroll
        for (int nc = 0; nc < 8; ++nc) {
            FragAB bf;
            const int bn = nc * 16 + lrow;
            #pragma unroll
            for (int j = 0; j < 8; ++j) {
                int kb = ((j & 4) << 2) + hi * 8 + (j & 3) * 2;
                bf.u[j] = *(const unsigned int*)&lds_bt[bn][kb];
            }
            acc[nc] = __builtin_amdgcn_wmma_f32_16x16x32_bf16(
                false, af.v, false, bf.v, (short)0, acc[nc], false, false);
        }
        __syncthreads();
    }

    // ---- epilogue: + bias, store ----
    #pragma unroll
    for (int nc = 0; nc < 8; ++nc) {
        #pragma unroll
        for (int r = 0; r < 8; ++r) {
            int gm = m0 + wave * 16 + r + hi * 8;
            int gn = n0 + nc * 16 + lrow;
            float val = acc[nc][r] + bias[gn];
            if (OUT_F32)
                ((float*)Cptr)[(size_t)gm * N + gn] = val;
            else
                ((unsigned short*)Cptr)[(size_t)gm * N + gn] = f32_to_bf16(val);
        }
    }
}

// ---------------------------------------------------------------------------
// Kernel 2: RoPE + scatter qkv[B,S,6144] -> q[B,NH,S,HD], k/v[B,NKV,S,HD]
// ---------------------------------------------------------------------------
__global__ __launch_bounds__(256)
void rope_scatter(const unsigned short* __restrict__ qkv,
                  const float* __restrict__ cosb,
                  const float* __restrict__ sinb,
                  unsigned short* __restrict__ q,
                  unsigned short* __restrict__ k,
                  unsigned short* __restrict__ v)
{
    const int bid = blockIdx.x;
    const int b   = bid / (S_ * NKV_);
    const int rem = bid % (S_ * NKV_);
    const int s   = rem / NKV_;
    const int kv  = rem % NKV_;

    const size_t base = ((size_t)b * S_ + s) * NQKV + (size_t)kv * (G_ * HD_ + 2 * HD_);

    #pragma unroll
    for (int it = 0; it < 3; ++it) {
        int idx = it * 256 + threadIdx.x; // 0..767
        if (idx < G_ * HD_) {             // ---- Q heads ----
            int hl = idx >> 7;            // local head
            int d  = idx & 127;
            int d2 = (d < HD_ / 2) ? d + HD_ / 2 : d - HD_ / 2;
            float x  = bf16_to_f32(qkv[base + idx]);
            float xr = bf16_to_f32(qkv[base + (size_t)hl * HD_ + d2]);
            float rot = (d < HD_ / 2) ? -xr : xr;
            float val = x * cosb[(size_t)s * HD_ + d] + rot * sinb[(size_t)s * HD_ + d];
            int gh = kv * G_ + hl;
            q[(((size_t)b * NH_ + gh) * S_ + s) * HD_ + d] = f32_to_bf16(val);
        } else if (idx < G_ * HD_ + HD_) { // ---- K head ----
            int d  = idx - G_ * HD_;
            int d2 = (d < HD_ / 2) ? d + HD_ / 2 : d - HD_ / 2;
            float x  = bf16_to_f32(qkv[base + idx]);
            float xr = bf16_to_f32(qkv[base + G_ * HD_ + d2]);
            float rot = (d < HD_ / 2) ? -xr : xr;
            float val = x * cosb[(size_t)s * HD_ + d] + rot * sinb[(size_t)s * HD_ + d];
            k[(((size_t)b * NKV_ + kv) * S_ + s) * HD_ + d] = f32_to_bf16(val);
        } else {                           // ---- V head (copy) ----
            int d = idx - (G_ * HD_ + HD_);
            v[(((size_t)b * NKV_ + kv) * S_ + s) * HD_ + d] = qkv[base + idx];
        }
    }
}

// ---------------------------------------------------------------------------
// TDM descriptor builders (Tensor Data Mover, gfx1250)
// 2D tile: 128 bf16 per row x 64 rows, row stride 128 elems, LDS row pad
// of 4 DWORDs (=> padded [64][136] ushort layout, bank-conflict free).
// ---------------------------------------------------------------------------
__device__ __forceinline__ u32x4 tdm_group0(unsigned lds_off, unsigned long long gaddr) {
    u32x4 g;
    g[0] = 1u;                              // count=1 (valid descriptor)
    g[1] = lds_off;                         // lds_addr (bytes)
    g[2] = (unsigned)(gaddr & 0xFFFFFFFFu); // global_addr[31:0]
    g[3] = (unsigned)((gaddr >> 32) & 0x1FFFFFFu) | 0x80000000u; // addr[56:32] | type=2
    return g;
}
__device__ __forceinline__ i32x8 tdm_group1_kv() {
    i32x8 g;
    // data_size=1 (2B) | pad_enable | pad_interval=5 (64 DW = 256B rows) |
    // pad_amount=3 (4 DWORDs = 16B pad per row)
    g[0] = (1 << 16) | (1 << 20) | (5 << 22) | (3 << 25);
    g[1] = (HD_ & 0xFFFF) << 16;            // tensor_dim0[15:0] = 128
    g[2] = (64 & 0xFFFF) << 16;             // tensor_dim0 hi=0 | tensor_dim1 = 64
    g[3] = (HD_ & 0xFFFF) << 16;            // tensor_dim1 hi=0 | tile_dim0 = 128
    g[4] = 64;                              // tile_dim1 = 64 rows, tile_dim2 = 0
    g[5] = HD_;                             // tensor_dim0_stride = 128 elems
    g[6] = 0;                               // stride hi / tensor_dim1_stride lo
    g[7] = 0;
    return g;
}

// ---------------------------------------------------------------------------
// Kernel 3: flash attention (causal, GQA). Block = (b, h, 64 q-rows),
// 128 threads = 4 waves, each wave owns 16 q-rows and full HD=128 output.
// K/V tiles staged into LDS by the Tensor Data Mover.
// ---------------------------------------------------------------------------
__global__ __launch_bounds__(128)
void flash_attn(const unsigned short* __restrict__ q,
                const unsigned short* __restrict__ k,
                const unsigned short* __restrict__ v,
                unsigned short* __restrict__ out)
{
    __shared__ unsigned short lds_k[64][136];     // [key][d], 16B padded rows
    __shared__ unsigned short lds_v[64][136];
    __shared__ unsigned short lds_p[4][16][72];   // per-wave prob bounce

    const int tid  = threadIdx.x;
    const int wave = tid >> 5;
    const int lane = tid & 31;
    const int lrow = lane & 15;
    const int hi   = lane >> 4;

    const int qblk = blockIdx.x;       // 64-row q tile
    const int h    = blockIdx.y;
    const int b    = blockIdx.z;
    const int kvh  = h / G_;
    const int qrow0 = qblk * 64 + wave * 16;

    // LDS byte offsets for TDM descriptors (ptrtoint escapes the arrays)
    const unsigned ldsk_off = (unsigned)(unsigned long long)(void*)&lds_k[0][0];
    const unsigned ldsv_off = (unsigned)(unsigned long long)(void*)&lds_v[0][0];

    // ---- load Q fragments straight from global (A-frag layout) ----
    const size_t qbase = (((size_t)b * NH_ + h) * S_ + qrow0 + lrow) * HD_;
    FragAB qf[4];
    #pragma unroll
    for (int dc = 0; dc < 4; ++dc)
        #pragma unroll
        for (int j = 0; j < 8; ++j) {
            int kb = dc * 32 + ((j & 4) << 2) + hi * 8 + (j & 3) * 2;
            qf[dc].u[j] = *(const unsigned int*)(q + qbase + kb);
        }

    v8f   o[8];
    float mi[8], li[8];
    #pragma unroll
    for (int i = 0; i < 8; ++i) { o[i] = {}; mi[i] = -3.0e38f; li[i] = 0.f; }

    const size_t kbase = ((size_t)b * NKV_ + kvh) * S_ * HD_;
    const i32x8 g1 = tdm_group1_kv();
    const i32x4 gz4 = {0, 0, 0, 0};
    const i32x8 gz8 = {0, 0, 0, 0, 0, 0, 0, 0};
    const int nTiles = qblk + 1; // causal: key tiles 0..qblk

    for (int t = 0; t < nTiles; ++t) {
        __syncthreads(); // previous tile fully consumed before refill
        const int key0 = t * 64;

        // ---- TDM: DMA K and V 64x128 bf16 tiles into padded LDS ----
        if (wave == 0) {
            unsigned long long gk = (unsigned long long)(const void*)
                                    (k + kbase + (size_t)key0 * HD_);
            unsigned long long gv = (unsigned long long)(const void*)
                                    (v + kbase + (size_t)key0 * HD_);
            __builtin_amdgcn_tensor_load_to_lds(tdm_group0(ldsk_off, gk), g1,
                                                gz4, gz4, gz8, 0);
            __builtin_amdgcn_tensor_load_to_lds(tdm_group0(ldsv_off, gv), g1,
                                                gz4, gz4, gz8, 0);
            __builtin_amdgcn_s_wait_tensorcnt(0);
        }
        asm volatile("" ::: "memory");
        __syncthreads();

        // ---- scores: S = Q * K^T (16x64 per wave) ----
        v8f sacc[4];
        #pragma unroll
        for (int nc = 0; nc < 4; ++nc) {
            v8f a = {};
            const int keyc = nc * 16 + lrow;
            #pragma unroll
            for (int dc = 0; dc < 4; ++dc) {
                FragAB bf;
                #pragma unroll
                for (int j = 0; j < 8; ++j) {
                    int d = dc * 32 + ((j & 4) << 2) + hi * 8 + (j & 3) * 2;
                    bf.u[j] = *(const unsigned int*)&lds_k[keyc][d];
                }
                a = __builtin_amdgcn_wmma_f32_16x16x32_bf16(
                    false, qf[dc].v, false, bf.v, (short)0, a, false, false);
            }
            sacc[nc] = a;
        }

        // ---- scale + causal mask ----
        #pragma unroll
        for (int nc = 0; nc < 4; ++nc)
            #pragma unroll
            for (int r = 0; r < 8; ++r) {
                int key = key0 + nc * 16 + lrow;
                int qr  = qrow0 + r + hi * 8;
                sacc[nc][r] = sacc[nc][r] * SCALE_ + ((key > qr) ? NEGV : 0.f);
            }

        // ---- row max (across 16 lanes of each half-wave) ----
        float tm[8];
        #pragma unroll
        for (int r = 0; r < 8; ++r) {
            float m = fmaxf(fmaxf(sacc[0][r], sacc[1][r]),
                            fmaxf(sacc[2][r], sacc[3][r]));
            m = fmaxf(m, __shfl_xor(m, 1, 32));
            m = fmaxf(m, __shfl_xor(m, 2, 32));
            m = fmaxf(m, __shfl_xor(m, 4, 32));
            m = fmaxf(m, __shfl_xor(m, 8, 32));
            tm[r] = m;
        }

        float mn[8], al[8], rs[8];
        #pragma unroll
        for (int r = 0; r < 8; ++r) {
            mn[r] = fmaxf(mi[r], tm[r]);
            al[r] = __expf(mi[r] - mn[r]);
            rs[r] = 0.f;
        }

        // ---- probabilities -> LDS bounce (re-layout C-frag -> A-frag) ----
        #pragma unroll
        for (int nc = 0; nc < 4; ++nc)
            #pragma unroll
            for (int r = 0; r < 8; ++r) {
                float p = __expf(sacc[nc][r] - mn[r]);
                rs[r] += p;
                lds_p[wave][r + hi * 8][nc * 16 + lrow] = f32_to_bf16(p);
            }

        #pragma unroll
        for (int r = 0; r < 8; ++r) {
            float s = rs[r];
            s += __shfl_xor(s, 1, 32);
            s += __shfl_xor(s, 2, 32);
            s += __shfl_xor(s, 4, 32);
            s += __shfl_xor(s, 8, 32);
            li[r] = li[r] * al[r] + s;
            mi[r] = mn[r];
        }

        // wave-local DS completion (only this wave reads its lds_p patch)
        asm volatile("s_wait_dscnt 0" ::: "memory");

        FragAB pf[2];
        #pragma unroll
        for (int kc = 0; kc < 2; ++kc)
            #pragma unroll
            for (int j = 0; j < 8; ++j) {
                int kb = kc * 32 + ((j & 4) << 2) + hi * 8 + (j & 3) * 2;
                pf[kc].u[j] = *(const unsigned int*)&lds_p[wave][lrow][kb];
            }

        // ---- O = O*alpha + P * V ----
        #pragma unroll
        for (int dcol = 0; dcol < 8; ++dcol) {
            #pragma unroll
            for (int r = 0; r < 8; ++r) o[dcol][r] *= al[r];
            #pragma unroll
            for (int kc = 0; kc < 2; ++kc) {
                FragAB vf;
                const int d = dcol * 16 + lrow;
                #pragma unroll
                for (int j = 0; j < 8; ++j) {
                    int key = kc * 32 + ((j & 4) << 2) + hi * 8 + (j & 3) * 2;
                    vf.s[2 * j]     = lds_v[key][d];
                    vf.s[2 * j + 1] = lds_v[key + 1][d];
                }
                o[dcol] = __builtin_amdgcn_wmma_f32_16x16x32_bf16(
                    false, pf[kc].v, false, vf.v, (short)0, o[dcol], false, false);
            }
        }
    }

    // ---- normalize + store: out[b, s, h*HD + d] (bf16) ----
    float inv[8];
    #pragma unroll
    for (int r = 0; r < 8; ++r) inv[r] = 1.0f / li[r];
    #pragma unroll
    for (int dcol = 0; dcol < 8; ++dcol)
        #pragma unroll
        for (int r = 0; r < 8; ++r) {
            int qr = qrow0 + r + hi * 8;
            int c  = h * HD_ + dcol * 16 + lrow;
            out[((size_t)b * S_ + qr) * H_ + c] = f32_to_bf16(o[dcol][r] * inv[r]);
        }
}

// ---------------------------------------------------------------------------
// Host launcher
// ---------------------------------------------------------------------------
extern "C" void kernel_launch(void* const* d_in, const int* in_sizes, int n_in,
                              void* d_out, int out_size, void* d_ws, size_t ws_size,
                              hipStream_t stream) {
    const float* hidden   = (const float*)d_in[0];
    // d_in[1] = attention_mask (pure causal; handled analytically in-kernel)
    const float* rope_cos = (const float*)d_in[2];
    const float* rope_sin = (const float*)d_in[3];
    const float* c_attn_w = (const float*)d_in[4];
    const float* c_attn_b = (const float*)d_in[5];
    const float* c_proj_w = (const float*)d_in[6];
    const float* c_proj_b = (const float*)d_in[7];
    float* outp = (float*)d_out;

    char* ws = (char*)d_ws;
    const size_t sz_qkv = (size_t)B_ * S_ * NQKV * 2;        // bf16
    const size_t sz_q   = (size_t)B_ * NH_  * S_ * HD_ * 2;  // bf16
    const size_t sz_kv  = (size_t)B_ * NKV_ * S_ * HD_ * 2;  // bf16
    unsigned short* qkv = (unsigned short*)(ws);
    unsigned short* qb  = (unsigned short*)(ws + sz_qkv);
    unsigned short* kb  = (unsigned short*)(ws + sz_qkv + sz_q);
    unsigned short* vb  = (unsigned short*)(ws + sz_qkv + sz_q + sz_kv);
    unsigned short* ao  = (unsigned short*)(ws + sz_qkv + sz_q + 2 * sz_kv);

    const int M = B_ * S_; // 4096

    // 1) QKV projection (fp32 A -> bf16, bf16 out)
    gemm_wmma_bf16<true, false>
        <<<dim3(NQKV / 128, M / 128), 256, 0, stream>>>(
            hidden, c_attn_w, c_attn_b, qkv, M, NQKV, H_);

    // 2) RoPE + split into q / k / v
    rope_scatter<<<B_ * S_ * NKV_, 256, 0, stream>>>(
        qkv, rope_cos, rope_sin, qb, kb, vb);

    // 3) Flash attention (causal, GQA) with TDM-staged K/V
    flash_attn<<<dim3(S_ / 64, NH_, B_), 128, 0, stream>>>(qb, kb, vb, ao);

    // 4) Output projection (bf16 A, fp32 out)
    gemm_wmma_bf16<false, true>
        <<<dim3(H_ / 128, M / 128), 256, 0, stream>>>(
            ao, c_proj_w, c_proj_b, outp, M, H_, H_);
}
